// TotalNeuralNetwork_52106543235382
// MI455X (gfx1250) — matrix operand order repaired and checked by reading, not compiled, not run on hardware
//
#include <hip/hip_runtime.h>
#include <hip/hip_bf16.h>

// ---------------------------------------------------------------------------
// Problem constants (match reference)
// ---------------------------------------------------------------------------
#define N_ATOMS 131072
#define N_ELEM  4
#define D_IN    128
#define D_HID   256

typedef __attribute__((ext_vector_type(16))) __bf16 v16bf;
typedef __attribute__((ext_vector_type(8)))  float  v8f;

__device__ __forceinline__ __bf16 f2bf(float f) {
    unsigned u = __float_as_uint(f);
    unsigned r = u + 0x7FFFu + ((u >> 16) & 1u);   // round-to-nearest-even
    unsigned short h = (unsigned short)(r >> 16);
    return __builtin_bit_cast(__bf16, h);
}

// Branchless tanh on hardware transcendentals:
//   tanh(x) = 1 - 2 / (exp2(2*log2(e)*x) + 1)
// Saturates correctly: x->+inf => exp2->inf => rcp->0 => 1 ; x->-inf => -1.
__device__ __forceinline__ float fast_tanh(float x) {
    float t = __builtin_amdgcn_exp2f(x * 2.8853900817779268f);
    float r = __builtin_amdgcn_rcpf(t + 1.0f);
    return __builtin_fmaf(r, -2.0f, 1.0f);
}

// ---------------------------------------------------------------------------
// Prep: W1 (E, D_IN, D_HID) f32  ->  W1T (E, D_HID, D_IN) bf16
// ---------------------------------------------------------------------------
__global__ void mlp_prep_w1t(const float* __restrict__ W1, __bf16* __restrict__ W1T) {
    int o = blockIdx.x * 256 + threadIdx.x;        // 0 .. E*D_HID*D_IN-1
    int e = o >> 15;                               // / (256*128)
    int h = (o >> 7) & 255;
    int i = o & 127;
    W1T[o] = f2bf(W1[e * (D_IN * D_HID) + i * D_HID + h]);
}

// ---------------------------------------------------------------------------
// Counting sort (deterministic, stable): 128 blocks x 1024 atoms
// ---------------------------------------------------------------------------
__global__ void mlp_hist(const int* __restrict__ ids, int* __restrict__ blockCounts) {
    __shared__ int sc[N_ELEM];
    int t = threadIdx.x;
    if (t < N_ELEM) sc[t] = 0;
    __syncthreads();
    int base = blockIdx.x * 1024 + t * 4;
    int c[N_ELEM] = {0, 0, 0, 0};
    for (int j = 0; j < 4; ++j) c[ids[base + j]]++;
    for (int e = 0; e < N_ELEM; ++e) if (c[e]) atomicAdd(&sc[e], c[e]);
    __syncthreads();
    if (t < N_ELEM) blockCounts[blockIdx.x * N_ELEM + t] = sc[t];
}

__global__ void mlp_scan(const int* __restrict__ blockCounts,
                         int* __restrict__ blockOffsets, int* __restrict__ S) {
    int tot[N_ELEM] = {0, 0, 0, 0};
    for (int b = 0; b < 128; ++b)
        for (int e = 0; e < N_ELEM; ++e) tot[e] += blockCounts[b * N_ELEM + e];
    S[0] = 0;
    for (int e = 0; e < N_ELEM; ++e) S[e + 1] = S[e] + tot[e];
    int run[N_ELEM];
    for (int e = 0; e < N_ELEM; ++e) run[e] = S[e];
    for (int b = 0; b < 128; ++b)
        for (int e = 0; e < N_ELEM; ++e) {
            blockOffsets[b * N_ELEM + e] = run[e];
            run[e] += blockCounts[b * N_ELEM + e];
        }
}

__global__ void mlp_scatter(const int* __restrict__ ids,
                            const int* __restrict__ blockOffsets,
                            int* __restrict__ perm) {
    __shared__ int tc[256 * N_ELEM];
    int t = threadIdx.x;
    int base = blockIdx.x * 1024 + t * 4;
    int myid[4];
    int c[N_ELEM] = {0, 0, 0, 0};
    for (int j = 0; j < 4; ++j) { myid[j] = ids[base + j]; c[myid[j]]++; }
    for (int e = 0; e < N_ELEM; ++e) tc[t * N_ELEM + e] = c[e];
    __syncthreads();
    if (t == 0) {                                   // serial exclusive scan (stable)
        int run[N_ELEM] = {0, 0, 0, 0};
        for (int i = 0; i < 256; ++i)
            for (int e = 0; e < N_ELEM; ++e) {
                int v = tc[i * N_ELEM + e];
                tc[i * N_ELEM + e] = run[e];
                run[e] += v;
            }
    }
    __syncthreads();
    int pos[N_ELEM];
    for (int e = 0; e < N_ELEM; ++e)
        pos[e] = blockOffsets[blockIdx.x * N_ELEM + e] + tc[t * N_ELEM + e];
    for (int j = 0; j < 4; ++j) { int e = myid[j]; perm[pos[e]++] = base + j; }
}

// ---------------------------------------------------------------------------
// Main: one wave per 32-atom tile. bf16 WMMA 16x16x32, f32 accumulate.
//   A: 2 M-tiles x 4 K-chunks (gathered rows, f32->bf16)
//   B: W1T fragments (contiguous per lane); fused bias+tanh+layer2
// ---------------------------------------------------------------------------
#define TILES_MAX ((N_ATOMS / 32) + N_ELEM)        // 4100 = 1025 blocks * 4 waves

__global__ __launch_bounds__(128) void
mlp_forward(const float* __restrict__ desc, const float* __restrict__ b1,
            const float* __restrict__ W2, const float* __restrict__ b2,
            const __bf16* __restrict__ W1T, const int* __restrict__ perm,
            const int* __restrict__ S, float* __restrict__ partials) {
    const int lane = threadIdx.x & 31;
    const int gw   = blockIdx.x * 4 + (threadIdx.x >> 5);

    // map wave -> (expert, tile)
    int e = -1, rowBase = 0, end = 0;
    int tilesBefore = 0;
    for (int ee = 0; ee < N_ELEM; ++ee) {
        int s = S[ee], s1 = S[ee + 1];
        int nt = (s1 - s + 31) >> 5;
        if (gw < tilesBefore + nt) { e = ee; rowBase = s + (gw - tilesBefore) * 32; end = s1; break; }
        tilesBefore += nt;
    }
    if (e < 0) { if (lane == 0) partials[gw] = 0.0f; return; }

    const int m  = lane & 15;
    const int r0 = rowBase + m, r1 = rowBase + 16 + m;
    const int a0 = perm[r0 < end ? r0 : rowBase];
    const int a1 = perm[r1 < end ? r1 : rowBase];
    const float* row0 = desc + (size_t)a0 * D_IN;
    const float* row1 = desc + (size_t)a1 * D_IN;
    const int kk = (lane < 16) ? 0 : 16;           // A-layout half-wave K offset

    // Load + convert A fragments: A[c] covers K = [32c, 32c+32)
    v16bf A0[4], A1[4];
#pragma unroll
    for (int c = 0; c < 4; ++c) {
        int kb = c * 32 + kk;
#pragma unroll
        for (int g = 0; g < 2; ++g) {              // g=0 -> K kb..kb+7 ; g=1 -> kb+8..kb+15
            const float4* p0 = (const float4*)(row0 + kb + g * 8);
            const float4* p1 = (const float4*)(row1 + kb + g * 8);
            float4 x0 = p0[0], x1 = p0[1];
            float4 y0 = p1[0], y1 = p1[1];
            int b = 8 * g;
            A0[c][b + 0] = f2bf(x0.x); A0[c][b + 1] = f2bf(x0.y);
            A0[c][b + 2] = f2bf(x0.z); A0[c][b + 3] = f2bf(x0.w);
            A0[c][b + 4] = f2bf(x1.x); A0[c][b + 5] = f2bf(x1.y);
            A0[c][b + 6] = f2bf(x1.z); A0[c][b + 7] = f2bf(x1.w);
            A1[c][b + 0] = f2bf(y0.x); A1[c][b + 1] = f2bf(y0.y);
            A1[c][b + 2] = f2bf(y0.z); A1[c][b + 3] = f2bf(y0.w);
            A1[c][b + 4] = f2bf(y1.x); A1[c][b + 5] = f2bf(y1.y);
            A1[c][b + 6] = f2bf(y1.z); A1[c][b + 7] = f2bf(y1.w);
        }
    }

    float ys0[8], ys1[8];
#pragma unroll
    for (int r = 0; r < 8; ++r) { ys0[r] = 0.0f; ys1[r] = 0.0f; }

    const int bkk = (lane < 16) ? 0 : 16;          // B-layout half-wave K offset
#pragma unroll 2
    for (int nt = 0; nt < D_HID / 16; ++nt) {
        const int n = nt * 16 + m;                 // this lane's N column
        const float b1v = b1[e * D_HID + n];
        const float w2v = W2[e * D_HID + n];
        const __bf16* bcol = W1T + ((size_t)(e * D_HID + n)) * D_IN + bkk;

        v16bf Bv[4];
#pragma unroll
        for (int c = 0; c < 4; ++c) Bv[c] = *(const v16bf*)(bcol + c * 32);

        v8f acc0 = {}, acc1 = {};
#pragma unroll
        for (int c = 0; c < 4; ++c) {
            acc0 = __builtin_amdgcn_wmma_f32_16x16x32_bf16(
                false, A0[c], false, Bv[c], (short)0, acc0, false, false);
            acc1 = __builtin_amdgcn_wmma_f32_16x16x32_bf16(
                false, A1[c], false, Bv[c], (short)0, acc1, false, false);
        }
#pragma unroll
        for (int r = 0; r < 8; ++r) {
            ys0[r] += fast_tanh(acc0[r] + b1v) * w2v;
            ys1[r] += fast_tanh(acc1[r] + b1v) * w2v;
        }
    }

    // masked per-lane sum over this lane's M rows, then full-wave reduce
    const int moff = (lane < 16) ? 0 : 8;
    float s = 0.0f;
#pragma unroll
    for (int r = 0; r < 8; ++r) {
        if (rowBase + (r + moff) < end)      s += ys0[r];
        if (rowBase + 16 + (r + moff) < end) s += ys1[r];
    }
#pragma unroll
    for (int off = 16; off > 0; off >>= 1) s += __shfl_xor(s, off, 32);

    int va = end - rowBase; if (va > 32) va = 32;  // valid atoms in tile (for b2)
    if (lane == 0) partials[gw] = s + b2[e] * (float)va;
}

// ---------------------------------------------------------------------------
// Final deterministic reduction
// ---------------------------------------------------------------------------
__global__ void mlp_reduce(const float* __restrict__ partials, int n, float* __restrict__ out) {
    __shared__ float sm[256];
    float s = 0.0f;
    for (int i = threadIdx.x; i < n; i += 256) s += partials[i];
    sm[threadIdx.x] = s;
    __syncthreads();
    for (int off = 128; off > 0; off >>= 1) {
        if (threadIdx.x < (unsigned)off) sm[threadIdx.x] += sm[threadIdx.x + off];
        __syncthreads();
    }
    if (threadIdx.x == 0) out[0] = sm[0];
}

// ---------------------------------------------------------------------------
extern "C" void kernel_launch(void* const* d_in, const int* in_sizes, int n_in,
                              void* d_out, int out_size, void* d_ws, size_t ws_size,
                              hipStream_t stream) {
    const float* desc = (const float*)d_in[0];
    const int*   ids  = (const int*)d_in[1];
    const float* W1   = (const float*)d_in[2];
    const float* b1   = (const float*)d_in[3];
    const float* W2   = (const float*)d_in[4];
    const float* b2   = (const float*)d_in[5];
    float* out = (float*)d_out;

    char* ws = (char*)d_ws;
    int*    perm         = (int*)(ws + 0);                       // 512 KB
    __bf16* W1T          = (__bf16*)(ws + 524288);               // 256 KB
    int*    blockCounts  = (int*)(ws + 786432);                  // 2 KB
    int*    blockOffsets = (int*)(ws + 788480);                  // 2 KB
    int*    S            = (int*)(ws + 790528);                  // 32 B
    float*  partials     = (float*)(ws + 790784);                // 16.4 KB

    mlp_prep_w1t<<<(N_ELEM * D_IN * D_HID) / 256, 256, 0, stream>>>(W1, W1T);
    mlp_hist<<<N_ATOMS / 1024, 256, 0, stream>>>(ids, blockCounts);
    mlp_scan<<<1, 1, 0, stream>>>(blockCounts, blockOffsets, S);
    mlp_scatter<<<N_ATOMS / 1024, 256, 0, stream>>>(ids, blockOffsets, perm);
    mlp_forward<<<TILES_MAX / 4, 128, 0, stream>>>(desc, b1, W2, b2, W1T, perm, S, partials);
    mlp_reduce<<<1, 256, 0, stream>>>(partials, TILES_MAX, out);
}
